// Block_6631429505693
// MI455X (gfx1250) — compile-verified, matrix operand order checked by reading, not run on hardware
//
#include <hip/hip_runtime.h>

// ---------------------------------------------------------------------------
// Transformer block (B=8, T=2048, C=384, H=6, HS=64) for gfx1250 / MI455X.
// All GEMMs use V_WMMA_F32_16X16X32_BF16 (bf16 A/B, f32 accumulate).
// Weights/activations are pre-converted to bf16 and pre-swizzled so the hot
// loops are pure aligned-dword loads + WMMA (no per-element cvt in the loop).
// Attention is flash-style (online softmax, 32-key blocks, no TxT in memory).
// ---------------------------------------------------------------------------

typedef __attribute__((ext_vector_type(16))) __bf16 v16bf;
typedef __attribute__((ext_vector_type(8)))  float  v8f;

union BF16x16 {
    v16bf v;
    unsigned int w[8];
    unsigned short h[16];
};

#define TB 8
#define TT 2048
#define TC 384
#define TH 6
#define THS 64
#define TBT (TB * TT)      // 16384 rows
#define TFF (4 * TC)       // 1536

// round-to-nearest-even f32 -> bf16
static __device__ __forceinline__ unsigned short f32_to_bf16(float f) {
    unsigned int u = __float_as_uint(f);
    u += 0x7FFFu + ((u >> 16) & 1u);
    return (unsigned short)(u >> 16);
}

// K-pair base for A/B 16-bit WMMA operand layout (ISA 7.12.2):
// VGPR j holds K = {kb, kb+1} packed in [15:0]/[31:16]; lanes 16-31: K += 8.
static __device__ __forceinline__ int kpair_base(int j, int hl) {
    return ((j & 4) << 2) + ((j & 3) << 1) + (hl << 3);
}

// Tile from bf16 row-major source, K contiguous within a row: 8 dword loads.
static __device__ __forceinline__ BF16x16
ld_tile_bf16(const unsigned short* base, size_t row, int ldr, int k0, int hl) {
    BF16x16 r;
    const unsigned short* p = base + row * (size_t)ldr + k0;
#pragma unroll
    for (int j = 0; j < 8; ++j)
        r.w[j] = *(const unsigned int*)(p + kpair_base(j, hl));
    return r;
}

// B tile from pre-swizzled weight panel [K/32][N][32] bf16: 8 dword loads.
static __device__ __forceinline__ BF16x16
ld_tileB_sw(const unsigned short* Bs, int k0, size_t col, int N, int hl) {
    BF16x16 r;
    const unsigned short* p = Bs + ((size_t)(k0 >> 5) * N + col) * 32;
#pragma unroll
    for (int j = 0; j < 8; ++j)
        r.w[j] = *(const unsigned int*)(p + kpair_base(j, hl));
    return r;
}

static __device__ __forceinline__ v8f
wmma_bf16(const BF16x16& a, const BF16x16& b, v8f c) {
    return __builtin_amdgcn_wmma_f32_16x16x32_bf16(
        false, a.v, false, b.v, (short)0, c, false, false);
}

// ---------------------------------------------------------------------------
// 0a) elementwise fp32 -> bf16 (for x)
// ---------------------------------------------------------------------------
__global__ void __launch_bounds__(256)
cvt_bf16_kernel(const float* __restrict__ in, unsigned short* __restrict__ out,
                int n) {
    int i = blockIdx.x * 256 + threadIdx.x;
    if (i < n) out[i] = f32_to_bf16(in[i]);
}

// ---------------------------------------------------------------------------
// 0b) weight swizzle: W fp32 [K,N] (batched) -> bf16 [K/32][N][32] panels
// ---------------------------------------------------------------------------
__global__ void __launch_bounds__(256)
swizzleB_kernel(const float* __restrict__ W, unsigned short* __restrict__ out,
                int K, int N) {
    const size_t mat = (size_t)blockIdx.y * K * N;
    int i = blockIdx.x * 256 + threadIdx.x;
    if (i >= K * N) return;
    int k = i / N, n = i - k * N;
    out[mat + ((size_t)(k >> 5) * N + n) * 32 + (k & 31)] =
        f32_to_bf16(W[mat + i]);
}

// ---------------------------------------------------------------------------
// 1) QKV projection: Xb[BT,C] x Ws[h] -> Q,K [B,H,T,HS] bf16, Vt [B,H,HS,T]
//    grid (BT/16, H, 3), block 128 (4 waves, one 16-col tile of HS each)
// ---------------------------------------------------------------------------
__global__ void __launch_bounds__(128)
qkv_kernel(const unsigned short* __restrict__ xb,
           const unsigned short* __restrict__ wqs,
           const unsigned short* __restrict__ wks,
           const unsigned short* __restrict__ wvs,
           unsigned short* __restrict__ qbf, unsigned short* __restrict__ kbf,
           unsigned short* __restrict__ vtbf) {
    const int lane = threadIdx.x & 31;
    const int l16  = lane & 15;
    const int hl   = (lane >> 4) & 1;
    const int g    = threadIdx.x >> 5;        // HS 16-col tile
    const int row0 = blockIdx.x * 16;         // row in [0, B*T)
    const int h    = blockIdx.y;
    const int which = blockIdx.z;             // 0=Q 1=K 2=V

    const unsigned short* Ws =
        (which == 0 ? wqs : which == 1 ? wks : wvs) + (size_t)h * TC * THS;
    const int n0 = g * 16;

    v8f acc = {};
    for (int k0 = 0; k0 < TC; k0 += 32) {
        BF16x16 a = ld_tile_bf16(xb, (size_t)(row0 + l16), TC, k0, hl);
        BF16x16 b = ld_tileB_sw(Ws, k0, (size_t)(n0 + l16), THS, hl);
        acc = wmma_bf16(a, b, acc);
    }

    const int bidx = row0 / TT;
    const int t0   = row0 - bidx * TT;
    const size_t bh = (size_t)(bidx * TH + h);

    unsigned short* dst;
    size_t base;
    int strideM;
    if (which == 2) {
        dst = vtbf;
        base = (bh * THS + n0 + l16) * TT + t0 + 8 * hl;
        strideM = 1;
    } else {
        dst = which ? kbf : qbf;
        base = (bh * TT + t0 + 8 * hl) * THS + n0 + l16;
        strideM = THS;
    }
#pragma unroll
    for (int j = 0; j < 8; ++j)
        dst[base + (size_t)j * strideM] = f32_to_bf16(acc[j]);
}

// ---------------------------------------------------------------------------
// 2) Flash attention: per (b,h,16-query tile), stream 32-key blocks.
//    S = Q K^T via 4 WMMA, online softmax, P via LDS (bf16) -> A layout,
//    PV via 4 WMMA. grid (T/16, H, B), block 32 (one wave). Output bf16.
// ---------------------------------------------------------------------------
#define PSTR 40   // u16 stride; even -> dword-aligned rows
__global__ void __launch_bounds__(32)
attn_kernel(const unsigned short* __restrict__ qbf,
            const unsigned short* __restrict__ kbf,
            const unsigned short* __restrict__ vtbf,
            unsigned short* __restrict__ attn_out) {
    const int lane = threadIdx.x & 31;
    const int l16  = lane & 15;
    const int hl   = (lane >> 4) & 1;
    const int t0   = blockIdx.x * 16;
    const int h    = blockIdx.y;
    const int b    = blockIdx.z;
    const size_t bh = (size_t)(b * TH + h);

    const unsigned short* Q  = qbf  + bh * TT * THS;
    const unsigned short* Kk = kbf  + bh * TT * THS;
    const unsigned short* Vt = vtbf + bh * THS * TT;

    // Q A-tiles for K-dim [0,32) and [32,64) — reused across all key blocks
    const BF16x16 aq0 = ld_tile_bf16(Q, (size_t)(t0 + l16), THS, 0,  hl);
    const BF16x16 aq1 = ld_tile_bf16(Q, (size_t)(t0 + l16), THS, 32, hl);

    float mrow[8], lrow[8];
    v8f o0 = {}, o1 = {}, o2 = {}, o3 = {};
#pragma unroll
    for (int j = 0; j < 8; ++j) { mrow[j] = -1e30f; lrow[j] = 0.0f; }

    __shared__ unsigned short pLds[16 * PSTR];

    for (int s0 = 0; s0 <= t0 + 15; s0 += 32) {
        // ---- scores: two 16x16 C/D tiles, K over HS=64 (2 wmma each)
        v8f sa = {}, sb = {};
        {
            BF16x16 bk0 = ld_tile_bf16(Kk, (size_t)(s0 + l16), THS, 0,  hl);
            BF16x16 bk1 = ld_tile_bf16(Kk, (size_t)(s0 + l16), THS, 32, hl);
            sa = wmma_bf16(aq0, bk0, sa);
            sa = wmma_bf16(aq1, bk1, sa);
        }
        {
            BF16x16 bk0 = ld_tile_bf16(Kk, (size_t)(s0 + 16 + l16), THS, 0,  hl);
            BF16x16 bk1 = ld_tile_bf16(Kk, (size_t)(s0 + 16 + l16), THS, 32, hl);
            sb = wmma_bf16(aq0, bk0, sb);
            sb = wmma_bf16(aq1, bk1, sb);
        }

        // ---- online softmax (row stats via 16-lane shfl reductions)
#pragma unroll
        for (int j = 0; j < 8; ++j) {
            const int trow = t0 + j + 8 * hl;
            float va = sa[j] * 0.125f;                 // 1/sqrt(64)
            float vb = sb[j] * 0.125f;
            if (s0 + l16 > trow)      va = -1e30f;     // causal mask
            if (s0 + 16 + l16 > trow) vb = -1e30f;
            float bm = fmaxf(va, vb);
#pragma unroll
            for (int off = 8; off; off >>= 1)
                bm = fmaxf(bm, __shfl_xor(bm, off, 32));
            const float mnew  = fmaxf(mrow[j], bm);
            const float alpha = __expf(mrow[j] - mnew);
            const float pa = __expf(va - mnew);
            const float pb = __expf(vb - mnew);
            float ps = pa + pb;
#pragma unroll
            for (int off = 8; off; off >>= 1)
                ps += __shfl_xor(ps, off, 32);
            lrow[j] = lrow[j] * alpha + ps;
            mrow[j] = mnew;
            pLds[(j + 8 * hl) * PSTR + l16]      = f32_to_bf16(pa);
            pLds[(j + 8 * hl) * PSTR + 16 + l16] = f32_to_bf16(pb);
            o0[j] *= alpha; o1[j] *= alpha; o2[j] *= alpha; o3[j] *= alpha;
        }

        // wave-local LDS RAW: C/D-layout P was written, now reread in A layout
        asm volatile("s_wait_dscnt 0" ::: "memory");

        BF16x16 pA;
        {
            const unsigned short* prow = pLds + (size_t)l16 * PSTR;
#pragma unroll
            for (int j = 0; j < 8; ++j)
                pA.w[j] = *(const unsigned int*)(prow + kpair_base(j, hl));
        }

        // ---- O += P x V  (V stored transposed -> K-contiguous B tiles)
        {
            BF16x16 bv = ld_tile_bf16(Vt, (size_t)(0 + l16),  TT, s0, hl);
            o0 = wmma_bf16(pA, bv, o0);
        }
        {
            BF16x16 bv = ld_tile_bf16(Vt, (size_t)(16 + l16), TT, s0, hl);
            o1 = wmma_bf16(pA, bv, o1);
        }
        {
            BF16x16 bv = ld_tile_bf16(Vt, (size_t)(32 + l16), TT, s0, hl);
            o2 = wmma_bf16(pA, bv, o2);
        }
        {
            BF16x16 bv = ld_tile_bf16(Vt, (size_t)(48 + l16), TT, s0, hl);
            o3 = wmma_bf16(pA, bv, o3);
        }
        asm volatile("s_wait_dscnt 0" ::: "memory");   // WAR vs next block's P writes
    }

    // ---- normalize and write concat-head output [B,T,C] as bf16 (GEMM A)
#pragma unroll
    for (int j = 0; j < 8; ++j) {
        const float inv  = 1.0f / lrow[j];
        const int   trow = t0 + j + 8 * hl;
        const size_t base = ((size_t)b * TT + trow) * TC + h * THS + l16;
        attn_out[base + 0]  = f32_to_bf16(o0[j] * inv);
        attn_out[base + 16] = f32_to_bf16(o1[j] * inv);
        attn_out[base + 32] = f32_to_bf16(o2[j] * inv);
        attn_out[base + 48] = f32_to_bf16(o3[j] * inv);
    }
}

// ---------------------------------------------------------------------------
// 3) WMMA GEMM: out[M,N] = act(Abf[M,K] @ Bs[K,N] + bias + residual)
//    A bf16 row-major, B pre-swizzled bf16. grid (M/16, N/16), block 32.
// ---------------------------------------------------------------------------
__global__ void __launch_bounds__(32)
gemm_kernel(const unsigned short* __restrict__ Abf,
            const unsigned short* __restrict__ Bs,
            const float* __restrict__ bias, const float* __restrict__ residual,
            float* __restrict__ outF, unsigned short* __restrict__ outBf,
            int K, int N, int relu) {
    const int lane = threadIdx.x & 31;
    const int l16  = lane & 15;
    const int hl   = (lane >> 4) & 1;
    const int row0 = blockIdx.x * 16;
    const int n0   = blockIdx.y * 16;
    const size_t arow = (size_t)(row0 + l16);
    const size_t col  = (size_t)(n0 + l16);

    v8f acc = {};
    for (int k0 = 0; k0 < K; k0 += 32) {
        if (k0 + 32 < K)   // hint next B panel (lowers to global_prefetch_b8)
            __builtin_prefetch(Bs + ((size_t)((k0 >> 5) + 1) * N + col) * 32, 0, 3);
        BF16x16 a = ld_tile_bf16(Abf, arow, K, k0, hl);
        BF16x16 b = ld_tileB_sw(Bs, k0, col, N, hl);
        acc = wmma_bf16(a, b, acc);
    }

#pragma unroll
    for (int j = 0; j < 8; ++j) {
        const int row = row0 + j + 8 * hl;
        const size_t idx = (size_t)row * N + col;
        float v = acc[j];
        if (bias)     v += bias[col];
        if (residual) v += residual[idx];
        if (relu)     v = fmaxf(v, 0.0f);
        if (outF)  outF[idx]  = v;
        if (outBf) outBf[idx] = f32_to_bf16(v);
    }
}

// ---------------------------------------------------------------------------
// 4) LayerNorm over C=384: one row per 128-thread block, 3 elems per thread.
//    Optional bf16 copy of the output (feeds the next GEMM's A operand).
// ---------------------------------------------------------------------------
__global__ void __launch_bounds__(128)
ln_kernel(const float* __restrict__ y, const float* __restrict__ g,
          const float* __restrict__ be, float* __restrict__ outF,
          unsigned short* __restrict__ outBf) {
    const int tid = threadIdx.x;
    const size_t row = blockIdx.x;
    const float* p = y + row * TC;
    float v0 = p[tid], v1 = p[tid + 128], v2 = p[tid + 256];

    __shared__ float red[128];
    red[tid] = v0 + v1 + v2;
    __syncthreads();
#pragma unroll
    for (int off = 64; off; off >>= 1) {
        if (tid < off) red[tid] += red[tid + off];
        __syncthreads();
    }
    const float mu = red[0] * (1.0f / TC);
    __syncthreads();

    const float d0 = v0 - mu, d1 = v1 - mu, d2 = v2 - mu;
    red[tid] = d0 * d0 + d1 * d1 + d2 * d2;
    __syncthreads();
#pragma unroll
    for (int off = 64; off; off >>= 1) {
        if (tid < off) red[tid] += red[tid + off];
        __syncthreads();
    }
    const float rs = rsqrtf(red[0] * (1.0f / TC) + 1e-5f);

    const float r0 = d0 * rs * g[tid]       + be[tid];
    const float r1 = d1 * rs * g[tid + 128] + be[tid + 128];
    const float r2 = d2 * rs * g[tid + 256] + be[tid + 256];
    if (outF) {
        float* o = outF + row * TC;
        o[tid] = r0; o[tid + 128] = r1; o[tid + 256] = r2;
    }
    if (outBf) {
        unsigned short* o = outBf + row * TC;
        o[tid] = f32_to_bf16(r0);
        o[tid + 128] = f32_to_bf16(r1);
        o[tid + 256] = f32_to_bf16(r2);
    }
}

// ---------------------------------------------------------------------------
// Host-side orchestration
// ---------------------------------------------------------------------------
extern "C" void kernel_launch(void* const* d_in, const int* in_sizes, int n_in,
                              void* d_out, int out_size, void* d_ws, size_t ws_size,
                              hipStream_t stream) {
    (void)in_sizes; (void)n_in; (void)out_size; (void)ws_size;

    const float* x      = (const float*)d_in[0];
    const float* wq     = (const float*)d_in[1];
    const float* wk     = (const float*)d_in[2];
    const float* wv     = (const float*)d_in[3];
    const float* w_proj = (const float*)d_in[4];
    const float* b_proj = (const float*)d_in[5];
    const float* w1     = (const float*)d_in[6];
    const float* b1     = (const float*)d_in[7];
    const float* w2     = (const float*)d_in[8];
    const float* b2     = (const float*)d_in[9];
    const float* g1     = (const float*)d_in[10];
    const float* be1    = (const float*)d_in[11];
    const float* g2     = (const float*)d_in[12];
    const float* be2    = (const float*)d_in[13];
    float* out = (float*)d_out;

    // ---- workspace layout (bytes)
    unsigned char* ws = (unsigned char*)d_ws;
    const size_t NE  = (size_t)TBT * TC;          // 6,291,456 elems
    unsigned short* qbf  = (unsigned short*)(ws + 0);
    unsigned short* kbf  = (unsigned short*)(ws + NE * 2);
    unsigned short* vtbf = (unsigned short*)(ws + NE * 4);
    unsigned short* aob  = (unsigned short*)(ws + NE * 6);   // attn out bf16
    unsigned short* ffh  = (unsigned short*)(ws + 0);        // overlay (50.3MB)
    float* y12           = (float*)(ws + NE * 8);            // y1 then y2
    float* x1            = (float*)(ws + NE * 8 + NE * 4);
    unsigned short* x1b  = (unsigned short*)(ws + NE * 8 + NE * 8);
    unsigned short* xb   = (unsigned short*)(ws + NE * 8 + NE * 8 + NE * 2);
    unsigned char* wsw   = ws + NE * 8 + NE * 8 + NE * 4;    // swizzled weights
    unsigned short* wqs  = (unsigned short*)(wsw + 0);
    unsigned short* wks  = (unsigned short*)(wsw + 294912);
    unsigned short* wvs  = (unsigned short*)(wsw + 589824);
    unsigned short* wps  = (unsigned short*)(wsw + 884736);
    unsigned short* w1s  = (unsigned short*)(wsw + 1179648);
    unsigned short* w2s  = (unsigned short*)(wsw + 2359296); // end +1179648

    // ---- 0) one-time conversions / swizzles (bandwidth-trivial)
    cvt_bf16_kernel<<<dim3((NE + 255) / 256), 256, 0, stream>>>(x, xb, (int)NE);
    swizzleB_kernel<<<dim3((TC * THS + 255) / 256, TH), 256, 0, stream>>>(wq, wqs, TC, THS);
    swizzleB_kernel<<<dim3((TC * THS + 255) / 256, TH), 256, 0, stream>>>(wk, wks, TC, THS);
    swizzleB_kernel<<<dim3((TC * THS + 255) / 256, TH), 256, 0, stream>>>(wv, wvs, TC, THS);
    swizzleB_kernel<<<dim3((TC * TC + 255) / 256, 1), 256, 0, stream>>>(w_proj, wps, TC, TC);
    swizzleB_kernel<<<dim3((TC * TFF + 255) / 256, 1), 256, 0, stream>>>(w1, w1s, TC, TFF);
    swizzleB_kernel<<<dim3((TFF * TC + 255) / 256, 1), 256, 0, stream>>>(w2, w2s, TFF, TC);

    // ---- 1) QKV projections (WMMA bf16)
    qkv_kernel<<<dim3(TBT / 16, TH, 3), 128, 0, stream>>>(
        xb, wqs, wks, wvs, qbf, kbf, vtbf);

    // ---- 2) flash attention (WMMA bf16, online softmax) -> bf16
    attn_kernel<<<dim3(TT / 16, TH, TB), 32, 0, stream>>>(qbf, kbf, vtbf, aob);

    // ---- 3) output projection + bias + residual(x) -> y1
    gemm_kernel<<<dim3(TBT / 16, TC / 16), 32, 0, stream>>>(
        aob, wps, b_proj, x, y12, nullptr, TC, TC, 0);

    // ---- 4) LayerNorm 1 -> x1 (fp32 residual copy) + x1b (bf16 GEMM A)
    ln_kernel<<<dim3(TBT), 128, 0, stream>>>(y12, g1, be1, x1, x1b);

    // ---- 5) FFN up + ReLU -> ffh (bf16, feeds next WMMA directly)
    gemm_kernel<<<dim3(TBT / 16, TFF / 16), 32, 0, stream>>>(
        x1b, w1s, b1, nullptr, nullptr, ffh, TC, TFF, 1);

    // ---- 6) FFN down + bias + residual(x1) -> y2
    gemm_kernel<<<dim3(TBT / 16, TC / 16), 32, 0, stream>>>(
        ffh, w2s, b2, x1, y12, nullptr, TFF, TC, 0);

    // ---- 7) LayerNorm 2 -> out
    ln_kernel<<<dim3(TBT), 128, 0, stream>>>(y12, g2, be2, out, nullptr);
}
